// ApexMambaOracle_84267258347639
// MI455X (gfx1250) — matrix operand order repaired and checked by reading, not compiled, not run on hardware
//
#include <hip/hip_runtime.h>

// ---------------------------------------------------------------------------
// ApexMambaOracle for MI455X (gfx1250)
// bf16 WMMA GEMMs (double-buffered LDS, peeled pipeline) + wave32 scan
// ---------------------------------------------------------------------------

#define N_ASSETS 2000
#define D_MODEL  1024
#define D_INNER  2048
#define D_STATE  128
#define D_CONV   4
#define DT_RANK  64
#define N_LAYERS 3
#define BATCH    2
#define SEQLEN   1024
#define BL       (BATCH * SEQLEN)   // 2048 rows for the main GEMMs

typedef __attribute__((ext_vector_type(16))) __bf16 v16bf;
typedef __attribute__((ext_vector_type(8)))  __bf16 v8bf;
typedef __attribute__((ext_vector_type(8)))  float  v8f;

// round-to-nearest-even fp32 -> bf16 (as raw u16)
__device__ __forceinline__ unsigned short f2bf(float f) {
  union { float f; unsigned int u; } v; v.f = f;
  unsigned int r = v.u + 0x7FFFu + ((v.u >> 16) & 1u);
  return (unsigned short)(r >> 16);
}

__device__ __forceinline__ float silu_f(float x) {
  return x / (1.0f + __expf(-x));
}

// ---------------------------------------------------------------------------
// fp32 -> bf16 bulk conversion (n must be a multiple of 4; all our sizes are)
// ---------------------------------------------------------------------------
__global__ __launch_bounds__(256)
void convert_f32_bf16(const float* __restrict__ src,
                      unsigned short* __restrict__ dst, long long n4)
{
  long long i = (long long)blockIdx.x * blockDim.x + threadIdx.x;
  if (i >= n4) return;
  float4 f = *(const float4*)(src + i * 4);
  union { unsigned short s[4]; uint2 u; } o;
  o.s[0] = f2bf(f.x); o.s[1] = f2bf(f.y);
  o.s[2] = f2bf(f.z); o.s[3] = f2bf(f.w);
  *(uint2*)(dst + i * 4) = o.u;
}

// ---------------------------------------------------------------------------
// NT GEMM, bf16 in / f32 accumulate:  C[M,N] = A[M,K] * W[N,K]^T (+bias[N])
// act==1: softplus.  Cf: optional fp32 output.  Cb: optional bf16 output.
// Block: 128x128 C-tile, 256 thr = 8 waves, wave -> 32(M) x 64(N) =
// 2x4 tiles of v_wmma_f32_16x16x32_bf16.  LDS double buffered, last k peeled.
// ---------------------------------------------------------------------------
#define TBM 128
#define TBN 128
#define TBK 32
#define LDSS 40   // bf16 per LDS row (32 + 8 pad, keeps 16B alignment)

__global__ __launch_bounds__(256)
void wmma_gemm_nt_bf16(const unsigned short* __restrict__ A, long long lda,
                       const unsigned short* __restrict__ W, long long ldw,
                       const float* __restrict__ bias,
                       float* __restrict__ Cf,
                       unsigned short* __restrict__ Cb, long long ldc,
                       int M, int N, int K, int act)
{
  __shared__ __align__(16) unsigned short smA[2][TBM * LDSS];
  __shared__ __align__(16) unsigned short smB[2][TBN * LDSS];

  const int tid   = threadIdx.x;
  const int lane  = tid & 31;
  const int wave  = tid >> 5;
  const int waveM = wave & 3;     // 0..3 -> 32-row slab
  const int waveN = wave >> 2;    // 0..1 -> 64-col slab
  const int m0 = blockIdx.y * TBM;
  const int n0 = blockIdx.x * TBN;

  v8f acc[2][4];
  #pragma unroll
  for (int mi = 0; mi < 2; ++mi)
    #pragma unroll
    for (int nj = 0; nj < 4; ++nj)
      #pragma unroll
      for (int e = 0; e < 8; ++e)
        acc[mi][nj][e] = 0.0f;

  // cooperative loader: each thread moves 16 contiguous bf16 per matrix/tile
  const int lr = tid >> 1;          // 0..127 (row within tile)
  const int lc = (tid & 1) * 16;    // 0 or 16 (K offset)
  const bool vA = (m0 + lr) < M;
  const bool vB = (n0 + lr) < N;
  const unsigned short* pA = A + (long long)(m0 + lr) * lda + lc;
  const unsigned short* pB = W + (long long)(n0 + lr) * ldw + lc;
  unsigned short* dA = &smA[0][lr * LDSS + lc];
  unsigned short* dB = &smB[0][lr * LDSS + lc];
  const int bufStride = TBM * LDSS;

  // fragment addressing (wave32)
  const int fr  = lane & 15;        // row within 16x16 tile
  const int fkA = (lane >> 4) * 16; // A: contiguous 16 K per lane half
  const int fkB = (lane >> 4) * 16; // B: contiguous 16 K per lane half

  // one k-tile of WMMA work from LDS buffer `buf`
  auto compute_tile = [&](int buf) {
    const unsigned short* bA = &smA[buf][0];
    const unsigned short* bB = &smB[buf][0];
    v16bf afrag[2];
    #pragma unroll
    for (int mi = 0; mi < 2; ++mi) {
      int r = waveM * 32 + mi * 16 + fr;
      v8bf lo = *(const v8bf*)&bA[r * LDSS + fkA];
      v8bf hi = *(const v8bf*)&bA[r * LDSS + fkA + 8];
      v16bf f;
      #pragma unroll
      for (int e = 0; e < 8; ++e) { f[e] = lo[e]; f[e + 8] = hi[e]; }
      afrag[mi] = f;
    }
    v16bf bfrag[4];
    #pragma unroll
    for (int nj = 0; nj < 4; ++nj) {
      int r = waveN * 64 + nj * 16 + fr;
      bfrag[nj] = *(const v16bf*)&bB[r * LDSS + fkB];
    }
    #pragma unroll
    for (int mi = 0; mi < 2; ++mi)
      #pragma unroll
      for (int nj = 0; nj < 4; ++nj)
        acc[mi][nj] = __builtin_amdgcn_wmma_f32_16x16x32_bf16(
            false, afrag[mi], false, bfrag[nj],
            (short)0, acc[mi][nj], false, false);
  };

  const int nk = K / TBK;
  uint4 ra0 = {0,0,0,0}, ra1 = {0,0,0,0}, rb0 = {0,0,0,0}, rb1 = {0,0,0,0};

  // prologue: tile 0 -> buf 0
  if (vA) { ra0 = *(const uint4*)pA; ra1 = *(const uint4*)(pA + 8); }
  if (vB) { rb0 = *(const uint4*)pB; rb1 = *(const uint4*)(pB + 8); }
  *(uint4*)dA = ra0; *(uint4*)(dA + 8) = ra1;
  *(uint4*)dB = rb0; *(uint4*)(dB + 8) = rb1;

  // steady state: unconditional next-tile load + compute + store
  for (int kt = 0; kt + 1 < nk; ++kt) {
    __syncthreads();
    const long long ko = (long long)(kt + 1) * TBK;
    ra0 = make_uint4(0,0,0,0); ra1 = ra0; rb0 = ra0; rb1 = ra0;
    if (vA) {
      ra0 = *(const uint4*)(pA + ko); ra1 = *(const uint4*)(pA + ko + 8);
      __builtin_prefetch(pA + ko + TBK, 0, 0);  // global_prefetch_b8
    }
    if (vB) {
      rb0 = *(const uint4*)(pB + ko); rb1 = *(const uint4*)(pB + ko + 8);
      __builtin_prefetch(pB + ko + TBK, 0, 0);
    }

    compute_tile(kt & 1);

    unsigned short* nA = dA + ((kt + 1) & 1) * bufStride;
    unsigned short* nB = dB + ((kt + 1) & 1) * bufStride;
    *(uint4*)nA = ra0; *(uint4*)(nA + 8) = ra1;
    *(uint4*)nB = rb0; *(uint4*)(nB + 8) = rb1;
  }
  // epilogue tile
  __syncthreads();
  compute_tile((nk - 1) & 1);

  // ---- epilogue: fold bias / softplus into acc, then store once ----
  const int rowOff = (lane >> 4) * 8;  // lanes 16..31 hold M rows +8
  const int colOff = lane & 15;
  #pragma unroll
  for (int nj = 0; nj < 4; ++nj) {
    int col = n0 + waveN * 64 + nj * 16 + colOff;
    float bv = (bias != nullptr && col < N) ? bias[col] : 0.0f;
    #pragma unroll
    for (int mi = 0; mi < 2; ++mi)
      #pragma unroll
      for (int v = 0; v < 8; ++v) {
        float val = acc[mi][nj][v] + bv;
        if (act == 1) val = (val > 20.0f) ? val : log1pf(__expf(val));
        acc[mi][nj][v] = val;
      }
  }
  if (Cf) {
    #pragma unroll
    for (int mi = 0; mi < 2; ++mi)
      #pragma unroll
      for (int nj = 0; nj < 4; ++nj) {
        int col = n0 + waveN * 64 + nj * 16 + colOff;
        #pragma unroll
        for (int v = 0; v < 8; ++v) {
          int row = m0 + waveM * 32 + mi * 16 + rowOff + v;
          if (row < M && col < N)
            Cf[(long long)row * ldc + col] = acc[mi][nj][v];
        }
      }
  }
  if (Cb) {
    #pragma unroll
    for (int mi = 0; mi < 2; ++mi)
      #pragma unroll
      for (int nj = 0; nj < 4; ++nj) {
        int col = n0 + waveN * 64 + nj * 16 + colOff;
        #pragma unroll
        for (int v = 0; v < 8; ++v) {
          int row = m0 + waveM * 32 + mi * 16 + rowOff + v;
          if (row < M && col < N)
            Cb[(long long)row * ldc + col] = f2bf(acc[mi][nj][v]);
        }
      }
  }
}

// ---------------------------------------------------------------------------
// Depthwise causal conv (D_CONV=4) over xb = xz[..., :D_INNER], +bias, silu.
// Emits fp32 (for the scan) and bf16 (for the W_x GEMM).
// ---------------------------------------------------------------------------
__global__ __launch_bounds__(256)
void conv_silu_kernel(const float* __restrict__ xz,        // [BL, 2*D_INNER]
                      const float* __restrict__ cw,        // [D_INNER, 4]
                      const float* __restrict__ cb,        // [D_INNER]
                      float* __restrict__ u,               // [BL, D_INNER]
                      unsigned short* __restrict__ ub)     // [BL, D_INNER] bf16
{
  long long idx = (long long)blockIdx.x * blockDim.x + threadIdx.x;
  if (idx >= (long long)BL * D_INNER) return;
  int d   = (int)(idx & (D_INNER - 1));
  int row = (int)(idx >> 11);          // b*L + t
  int t   = row & (SEQLEN - 1);
  const float* w = cw + d * 4;
  float acc = cb[d];
  #pragma unroll
  for (int j = 0; j < D_CONV; ++j) {
    int tt = t - (D_CONV - 1) + j;
    if (tt >= 0)
      acc += xz[(long long)(row - (D_CONV - 1) + j) * (2 * D_INNER) + d] * w[j];
  }
  float r = silu_f(acc);
  u[idx]  = r;
  ub[idx] = f2bf(r);
}

// ---------------------------------------------------------------------------
// Selective scan: one wave32 per (b, d) channel; 4 states per lane.
// Fuses y = (scan_out + u*Dp) * silu(z).  Output bf16 (feeds W_out GEMM).
// ---------------------------------------------------------------------------
__global__ __launch_bounds__(256)
void scan_kernel(const float* __restrict__ delta,  // [BL, D_INNER]
                 const float* __restrict__ u,      // [BL, D_INNER]
                 const float* __restrict__ xdbl,   // [BL, 320]; B at 64, C at 192
                 const float* __restrict__ xz,     // [BL, 2*D_INNER]; z at +D_INNER
                 const float* __restrict__ A_log,  // [D_INNER, D_STATE]
                 const float* __restrict__ Dp,     // [D_INNER]
                 unsigned short* __restrict__ y)   // [BL, D_INNER] bf16
{
  const int lane = threadIdx.x & 31;
  const int wave = threadIdx.x >> 5;
  const int ch   = blockIdx.x * 8 + wave;        // 0 .. B*D_INNER-1
  const int b    = ch >> 11;                     // / D_INNER
  const int d    = ch & (D_INNER - 1);

  float a0 = -__expf(A_log[(long long)d * D_STATE + lane +  0]);
  float a1 = -__expf(A_log[(long long)d * D_STATE + lane + 32]);
  float a2 = -__expf(A_log[(long long)d * D_STATE + lane + 64]);
  float a3 = -__expf(A_log[(long long)d * D_STATE + lane + 96]);
  const float dp = Dp[d];

  float s0 = 0.f, s1 = 0.f, s2 = 0.f, s3 = 0.f;

  for (int t = 0; t < SEQLEN; ++t) {
    const long long row = (long long)b * SEQLEN + t;
    const float dt = delta[row * D_INNER + d];
    const float ut = u[row * D_INNER + d];
    const float zt = xz[row * (2 * D_INNER) + D_INNER + d];
    const float* bc = xdbl + row * (DT_RANK + 2 * D_STATE);

    const float B0 = bc[DT_RANK + lane +  0], B1 = bc[DT_RANK + lane + 32];
    const float B2 = bc[DT_RANK + lane + 64], B3 = bc[DT_RANK + lane + 96];
    const float C0 = bc[DT_RANK + D_STATE + lane +  0], C1 = bc[DT_RANK + D_STATE + lane + 32];
    const float C2 = bc[DT_RANK + D_STATE + lane + 64], C3 = bc[DT_RANK + D_STATE + lane + 96];

    const float du = dt * ut;
    s0 = s0 * __expf(dt * a0) + du * B0;
    s1 = s1 * __expf(dt * a1) + du * B1;
    s2 = s2 * __expf(dt * a2) + du * B2;
    s3 = s3 * __expf(dt * a3) + du * B3;

    float p = s0 * C0 + s1 * C1 + s2 * C2 + s3 * C3;
    #pragma unroll
    for (int off = 16; off > 0; off >>= 1)
      p += __shfl_xor(p, off, 32);

    if (lane == 0)
      y[row * D_INNER + d] = f2bf((p + ut * dp) * silu_f(zt));
  }
}

// ---------------------------------------------------------------------------
// sigma[b,i,j] = sum_k fac[b,i*5+k]*fac[b,j*5+k] + (i==j)*1e-6
// ---------------------------------------------------------------------------
__global__ __launch_bounds__(256)
void sigma_kernel(const float* __restrict__ fac,   // [BATCH, N_ASSETS*5]
                  float* __restrict__ sigma)       // [BATCH, N_ASSETS, N_ASSETS]
{
  long long idx = (long long)blockIdx.x * blockDim.x + threadIdx.x;
  const long long total = (long long)BATCH * N_ASSETS * N_ASSETS;
  if (idx >= total) return;
  int b = (int)(idx / ((long long)N_ASSETS * N_ASSETS));
  int r = (int)(idx % ((long long)N_ASSETS * N_ASSETS));
  int i = r / N_ASSETS, j = r % N_ASSETS;
  const float* fi = fac + (long long)b * N_ASSETS * 5 + (long long)i * 5;
  const float* fj = fac + (long long)b * N_ASSETS * 5 + (long long)j * 5;
  float s = fi[0]*fj[0] + fi[1]*fj[1] + fi[2]*fj[2] + fi[3]*fj[3] + fi[4]*fj[4];
  if (i == j) s += 1e-6f;
  sigma[idx] = s;
}

// ---------------------------------------------------------------------------
// Host-side orchestration
// ---------------------------------------------------------------------------
static inline void launch_convert(hipStream_t stream, const float* src,
                                  unsigned short* dst, long long n)
{
  long long n4 = n / 4;
  convert_f32_bf16<<<dim3((unsigned)((n4 + 255) / 256)), 256, 0, stream>>>(src, dst, n4);
}

static inline void launch_gemm(hipStream_t stream,
                               const unsigned short* A, long long lda,
                               const unsigned short* W, long long ldw,
                               const float* bias,
                               float* Cf, unsigned short* Cb, long long ldc,
                               int M, int N, int K, int act)
{
  dim3 grid((N + TBN - 1) / TBN, (M + TBM - 1) / TBM);
  wmma_gemm_nt_bf16<<<grid, 256, 0, stream>>>(A, lda, W, ldw, bias, Cf, Cb, ldc,
                                              M, N, K, act);
}

extern "C" void kernel_launch(void* const* d_in, const int* in_sizes, int n_in,
                              void* d_out, int out_size, void* d_ws, size_t ws_size,
                              hipStream_t stream) {
  (void)in_sizes; (void)n_in; (void)out_size; (void)ws_size;

  const float* x      = (const float*)d_in[0];
  const float* W_proj = (const float*)d_in[1];
  const float* b_proj = (const float*)d_in[2];
  const float* W_in   = (const float*)d_in[3];
  const float* conv_w = (const float*)d_in[4];
  const float* conv_b = (const float*)d_in[5];
  const float* W_x    = (const float*)d_in[6];
  const float* W_dt   = (const float*)d_in[7];
  const float* b_dt   = (const float*)d_in[8];
  const float* A_log  = (const float*)d_in[9];
  const float* Dp     = (const float*)d_in[10];
  const float* W_out  = (const float*)d_in[11];
  const float* mu_W   = (const float*)d_in[12];
  const float* mu_b   = (const float*)d_in[13];
  const float* fac_W  = (const float*)d_in[14];
  const float* fac_b  = (const float*)d_in[15];

  float* out = (float*)d_out;                 // mu [2,2000] then sigma [2,2000,2000]

  // ---- workspace layout ----
  // fp32 region
  float* xz    = (float*)d_ws;                                    // [BL, 4096]
  float* u     = xz    + (long long)BL * 2 * D_INNER;             // [BL, 2048]
  float* xdbl  = u     + (long long)BL * D_INNER;                 // [BL, 320]
  float* delta = xdbl  + (long long)BL * (DT_RANK + 2 * D_STATE); // [BL, 2048]
  float* fac   = delta + (long long)BL * D_INNER;                 // [2, 10000]
  // bf16 region
  unsigned short* x_bf  = (unsigned short*)(fac + BATCH * N_ASSETS * 5);
  unsigned short* h_bf0 = x_bf  + (long long)BL * (N_ASSETS * 4); // [BL, 1024]
  unsigned short* h_bf1 = h_bf0 + (long long)BL * D_MODEL;
  unsigned short* wstg  = h_bf1 + (long long)BL * D_MODEL;        // weight staging
  // x_bf (16.4M elems) is dead after the input projection; recycle it:
  unsigned short* u_bf    = x_bf;                                 // [BL, 2048]
  unsigned short* y_bf    = u_bf + (long long)BL * D_INNER;       // [BL, 2048]
  unsigned short* xdbl_bf = y_bf + (long long)BL * D_INNER;       // [BL, 320]

  // 1) input projection: h0 = x @ W_proj^T + b_proj   (bf16 out only)
  launch_convert(stream, x, x_bf, (long long)BL * N_ASSETS * 4);
  launch_convert(stream, W_proj, wstg, (long long)D_MODEL * N_ASSETS * 4);
  launch_gemm(stream, x_bf, N_ASSETS * 4, wstg, N_ASSETS * 4, b_proj,
              nullptr, h_bf0, D_MODEL, BL, D_MODEL, N_ASSETS * 4, 0);

  unsigned short* hcur  = h_bf0;
  unsigned short* hnext = h_bf1;
  for (int l = 0; l < N_LAYERS; ++l) {
    const float* W_in_l   = W_in   + (long long)l * 2 * D_INNER * D_MODEL;
    const float* conv_w_l = conv_w + (long long)l * D_INNER * D_CONV;
    const float* conv_b_l = conv_b + (long long)l * D_INNER;
    const float* W_x_l    = W_x    + (long long)l * (DT_RANK + 2 * D_STATE) * D_INNER;
    const float* W_dt_l   = W_dt   + (long long)l * D_INNER * DT_RANK;
    const float* b_dt_l   = b_dt   + (long long)l * D_INNER;
    const float* A_log_l  = A_log  + (long long)l * D_INNER * D_STATE;
    const float* Dp_l     = Dp     + (long long)l * D_INNER;
    const float* W_out_l  = W_out  + (long long)l * D_MODEL * D_INNER;

    // xz = h @ W_in^T (fp32 out: feeds conv + gate)
    launch_convert(stream, W_in_l, wstg, (long long)2 * D_INNER * D_MODEL);
    launch_gemm(stream, hcur, D_MODEL, wstg, D_MODEL, nullptr,
                xz, nullptr, 2 * D_INNER, BL, 2 * D_INNER, D_MODEL, 0);

    // u = silu(causal depthwise conv(xb) + conv_b)  (fp32 + bf16)
    {
      long long n = (long long)BL * D_INNER;
      conv_silu_kernel<<<dim3((unsigned)((n + 255) / 256)), 256, 0, stream>>>(
          xz, conv_w_l, conv_b_l, u, u_bf);
    }

    // x_dbl = u @ W_x^T  (fp32 for scan B/C + bf16 for dt GEMM)
    launch_convert(stream, W_x_l, wstg, (long long)(DT_RANK + 2 * D_STATE) * D_INNER);
    launch_gemm(stream, u_bf, D_INNER, wstg, D_INNER, nullptr,
                xdbl, xdbl_bf, DT_RANK + 2 * D_STATE,
                BL, DT_RANK + 2 * D_STATE, D_INNER, 0);

    // delta = softplus(dt @ W_dt^T + b_dt)
    launch_convert(stream, W_dt_l, wstg, (long long)D_INNER * DT_RANK);
    launch_gemm(stream, xdbl_bf, DT_RANK + 2 * D_STATE, wstg, DT_RANK, b_dt_l,
                delta, nullptr, D_INNER, BL, D_INNER, DT_RANK, 1);

    // selective scan; fuses +u*Dp and *silu(z); emits bf16 y
    scan_kernel<<<dim3(BATCH * D_INNER / 8), 256, 0, stream>>>(
        delta, u, xdbl, xz, A_log_l, Dp_l, y_bf);

    // h_next = y @ W_out^T  (bf16 out only)
    launch_convert(stream, W_out_l, wstg, (long long)D_MODEL * D_INNER);
    launch_gemm(stream, y_bf, D_INNER, wstg, D_INNER, nullptr,
                nullptr, hnext, D_MODEL, BL, D_MODEL, D_INNER, 0);

    unsigned short* tmp = hcur; hcur = hnext; hnext = tmp;
  }

  // head: last = h[:, -1, :]  -> rows 1023 and 2047 of hcur
  const unsigned short* last = hcur + (long long)(SEQLEN - 1) * D_MODEL;
  const long long lastStride = (long long)SEQLEN * D_MODEL;

  // mu -> d_out[0 : 2*2000]
  launch_convert(stream, mu_W, wstg, (long long)N_ASSETS * D_MODEL);
  launch_gemm(stream, last, lastStride, wstg, D_MODEL, mu_b,
              out, nullptr, N_ASSETS, BATCH, N_ASSETS, D_MODEL, 0);

  // factors = last @ fac_W^T + fac_b
  launch_convert(stream, fac_W, wstg, (long long)N_ASSETS * 5 * D_MODEL);
  launch_gemm(stream, last, lastStride, wstg, D_MODEL, fac_b,
              fac, nullptr, N_ASSETS * 5, BATCH, N_ASSETS * 5, D_MODEL, 0);

  // sigma -> d_out[4000 : 4000 + 2*2000*2000]
  {
    long long n = (long long)BATCH * N_ASSETS * N_ASSETS;
    sigma_kernel<<<dim3((unsigned)((n + 255) / 256)), 256, 0, stream>>>(
        fac, out + (long long)BATCH * N_ASSETS);
  }
}
